// CausalEventModel_87892210745634
// MI455X (gfx1250) — matrix operand order repaired and checked by reading, not compiled, not run on hardware
//
#include <hip/hip_runtime.h>
#include <hip/hip_bf16.h>

// ---------------------------------------------------------------------------
// RWKV-ish model on MI455X (gfx1250).
// ~452 GFLOP of GEMM vs ~200MB working set @23.3TB/s -> compute bound.
// bf16 WMMA (v_wmma_f32_16x16x32_bf16) with f32 accumulate.
// GEMM staging uses CDNA5 async global->LDS copies (ASYNCcnt) with double
// buffering so tile t+1 copy overlaps tile t compute.
// WKV scan is serial over T (latency bound, 2048 channels) -> own kernel.
// ---------------------------------------------------------------------------

typedef __bf16 bf16;
typedef __attribute__((ext_vector_type(16))) __bf16 v16bf;
typedef __attribute__((ext_vector_type(8)))  float  v8f;
typedef __attribute__((ext_vector_type(4)))  unsigned int u32x4;

union FragBF { u32x4 u[2]; v16bf bf; };

#define BM 128
#define BN 128
#define BK 32
#define ABYTES (BM * BK * 2)   // 8 KB per A buffer
#define BBYTES (BN * BK * 2)   // 8 KB per B buffer

// epilogue flags
#define EPI_SIG    1
#define EPI_RELUSQ 2
#define EPI_ADD    4
#define EPI_GATE   8
#define EPI_BF16   16
#define EPI_F32    32

// CDNA5 async global->LDS copy, 16B per lane. vdst = LDS byte address,
// vaddr = 64-bit global address. Tracked by ASYNCcnt.
__device__ __forceinline__ void async_ld_b128(unsigned lds_addr, const void* gaddr) {
    asm volatile("global_load_async_to_lds_b128 %0, %1, off"
                 :: "v"(lds_addr), "v"(gaddr) : "memory");
}
__device__ __forceinline__ void wait_async0() {
    asm volatile("s_wait_asynccnt 0x0" ::: "memory");
}

// ---------------------------------------------------------------------------
// bf16 GEMM:  C[M,N](f32) (+)= epi( A_bf16[M,K] * Wt_bf16[N,K]^T )
// Weights pre-transposed to [N][K] so B tiles stage contiguously.
// 256 threads = 8 waves (4x2); wave tile 32x64 = 8 wmma per K-step.
// Double-buffered async staging: 32KB LDS.
// ---------------------------------------------------------------------------
__global__ __launch_bounds__(256) void gemm_bf16_kernel(
    const bf16* __restrict__ A, const bf16* __restrict__ Wt,
    float* __restrict__ C, bf16* __restrict__ Cbf,
    const float* __restrict__ gate,
    int M, int N, int K, int flags)
{
    __shared__ bf16 sA[2][BM * BK];
    __shared__ bf16 sB[2][BN * BK];

    const int tid  = threadIdx.x;
    const int bm   = blockIdx.y * BM;
    const int bn   = blockIdx.x * BN;
    const int wave = tid >> 5;
    const int lane = tid & 31;
    const int wm   = (wave >> 1) * 32;   // 0,32,64,96
    const int wn   = (wave & 1) * 64;    // 0,64
    const int half = lane >> 4;          // 0 or 1
    const int r    = lane & 15;

    // ---- per-thread staging descriptors: 2 x 16B chunks each for A and B ----
    // chunk c (0..511): row = c>>2 (4 x 16B per 32-elem row), q = c&3
    const int rA0 = tid >> 2,          qA0 = tid & 3;
    const int rA1 = (tid + 256) >> 2,  qA1 = tid & 3;   // (tid+256)&3 == tid&3
    const char* aSrc0 = (const char*)(A  + (size_t)(bm + rA0) * K + qA0 * 8);
    const char* aSrc1 = (const char*)(A  + (size_t)(bm + rA1) * K + qA1 * 8);
    const char* bSrc0 = (const char*)(Wt + (size_t)(bn + rA0) * K + qA0 * 8);
    const char* bSrc1 = (const char*)(Wt + (size_t)(bn + rA1) * K + qA1 * 8);

    const unsigned sAbase = (unsigned)(uintptr_t)&sA[0][0];
    const unsigned sBbase = (unsigned)(uintptr_t)&sB[0][0];
    const unsigned ldsA0 = sAbase + (unsigned)(rA0 * BK + qA0 * 8) * 2;
    const unsigned ldsA1 = sAbase + (unsigned)(rA1 * BK + qA1 * 8) * 2;
    const unsigned ldsB0 = sBbase + (unsigned)(rA0 * BK + qA0 * 8) * 2;
    const unsigned ldsB1 = sBbase + (unsigned)(rA1 * BK + qA1 * 8) * 2;

    auto issue_tile = [&](int buf, int t) {
        size_t koff = (size_t)t * (BK * 2);   // byte advance along K (row-major)
        unsigned ao = (unsigned)buf * ABYTES;
        unsigned bo = (unsigned)buf * BBYTES;
        async_ld_b128(ldsA0 + ao, aSrc0 + koff);
        async_ld_b128(ldsA1 + ao, aSrc1 + koff);
        async_ld_b128(ldsB0 + bo, bSrc0 + koff);
        async_ld_b128(ldsB1 + bo, bSrc1 + koff);
    };

    const int ntiles = K / BK;
    v8f acc[2][4] = {};

    issue_tile(0, 0);
    wait_async0();
    __syncthreads();

    for (int t = 0; t < ntiles; ++t) {
        const int buf = t & 1;
        if (t + 1 < ntiles) issue_tile(buf ^ 1, t + 1);   // overlap with compute

        const bf16* pA = &sA[buf][0];
        const bf16* pB = &sB[buf][0];

        // ---- gather fragments per ISA 7.12.2 layouts ----
        FragBF a[2], b[4];
        #pragma unroll
        for (int i = 0; i < 2; ++i) {
            int m = wm + i * 16 + r;
            // lanes 0-15: K 0..7 then 16..23 ; lanes 16-31: K 8..15 then 24..31
            a[i].u[0] = *(const u32x4*)(pA + m * BK + half * 8);
            a[i].u[1] = *(const u32x4*)(pA + m * BK + 16 + half * 8);
        }
        #pragma unroll
        for (int j = 0; j < 4; ++j) {
            int n = wn + j * 16 + r;
            // lane holds column n, K = half*8 .. half*8+7 then 16+half*8 .. 16+half*8+7
            b[j].u[0] = *(const u32x4*)(pB + n * BK + half * 8);
            b[j].u[1] = *(const u32x4*)(pB + n * BK + 16 + half * 8);
        }

        #pragma unroll
        for (int i = 0; i < 2; ++i)
            #pragma unroll
            for (int j = 0; j < 4; ++j)
                acc[i][j] = __builtin_amdgcn_wmma_f32_16x16x32_bf16(
                    false, a[i].bf, false, b[j].bf,
                    (short)0, acc[i][j], false, false);

        if (t + 1 < ntiles) {
            wait_async0();      // next buffer landed
            __syncthreads();    // ...and everyone finished reading current buf
        }
    }

    // ---- epilogue: C/D layout: VGPR v -> M = half*8 + v, N = lane%16 ----
    #pragma unroll
    for (int i = 0; i < 2; ++i) {
        #pragma unroll
        for (int j = 0; j < 4; ++j) {
            #pragma unroll
            for (int v = 0; v < 8; ++v) {
                int m = bm + wm + i * 16 + half * 8 + v;
                int n = bn + wn + j * 16 + r;
                size_t idx = (size_t)m * N + n;
                float val = acc[i][j][v];
                if (flags & EPI_SIG)    val = 1.0f / (1.0f + __expf(-val));
                if (flags & EPI_RELUSQ) { float tt = fmaxf(val, 0.0f); val = tt * tt; }
                if (flags & EPI_GATE)   val *= gate[idx];
                if (flags & EPI_ADD)       C[idx] += val;
                else if (flags & EPI_F32)  C[idx] = val;
                if (flags & EPI_BF16)      Cbf[idx] = (bf16)val;
            }
        }
    }
}

// ---------------------------------------------------------------------------
// WKV sequential scan: one thread per (b,d) channel. Writes r*wkv as bf16
// (A-operand for the Wo GEMM).
// ---------------------------------------------------------------------------
__global__ void wkv_kernel(const float* __restrict__ k, const float* __restrict__ v,
                           const float* __restrict__ rg,
                           const float* __restrict__ decay, const float* __restrict__ first,
                           bf16* __restrict__ rwkv_bf, int T, int D)
{
    int idx = blockIdx.x * blockDim.x + threadIdx.x;   // 0 .. B*D-1
    int b = idx / D, d = idx % D;
    float w = -__expf(decay[d]);
    float u = first[d];
    float aa = 0.f, bb = 0.f, pp = -1e30f;
    size_t base = (size_t)b * T * D + d;
    for (int t = 0; t < T; ++t) {
        size_t o = base + (size_t)t * D;
        float kt = k[o], vt = v[o];
        float ww = u + kt;
        float p  = fmaxf(pp, ww);
        float e1 = __expf(pp - p), e2 = __expf(ww - p);
        float out = (e1 * aa + e2 * vt) / (e1 * bb + e2);
        float ww2 = pp + w;
        float p2  = fmaxf(ww2, kt);
        float e1b = __expf(ww2 - p2), e2b = __expf(kt - p2);
        aa = e1b * aa + e2b * vt;
        bb = e1b * bb + e2b;
        pp = p2;
        rwkv_bf[o] = (bf16)(rg[o] * out);
    }
}

// ---------------------------------------------------------------------------
// Embedding (E=4) + input LayerNorm. One 256-thread block per token row.
// ---------------------------------------------------------------------------
__global__ __launch_bounds__(256) void embed_ln_kernel(
    const float* __restrict__ x, const float* __restrict__ emb_w,
    const float* __restrict__ emb_b,
    const float* __restrict__ lw, const float* __restrict__ lb,
    float* __restrict__ h, int E, int D)
{
    __shared__ float red[256];
    __shared__ float xe[8];
    int row = blockIdx.x, tid = threadIdx.x;
    if (tid < E) xe[tid] = x[(size_t)row * E + tid];
    __syncthreads();
    float v0, v1;
    {
        int d0 = tid, d1 = tid + 256;
        float s0 = emb_b[d0], s1 = emb_b[d1];
        for (int e = 0; e < E; ++e) {
            s0 += xe[e] * emb_w[e * D + d0];
            s1 += xe[e] * emb_w[e * D + d1];
        }
        v0 = s0; v1 = s1;
    }
    red[tid] = v0 + v1; __syncthreads();
    for (int s = 128; s > 0; s >>= 1) { if (tid < s) red[tid] += red[tid + s]; __syncthreads(); }
    float mean = red[0] / D;
    __syncthreads();
    float d0 = v0 - mean, d1 = v1 - mean;
    red[tid] = d0 * d0 + d1 * d1; __syncthreads();
    for (int s = 128; s > 0; s >>= 1) { if (tid < s) red[tid] += red[tid + s]; __syncthreads(); }
    float rstd = rsqrtf(red[0] / D + 1e-5f);
    float* op = h + (size_t)row * D;
    op[tid]       = d0 * rstd * lw[tid]       + lb[tid];
    op[tid + 256] = d1 * rstd * lw[tid + 256] + lb[tid + 256];
}

// LayerNorm over D=512, one block per row, 2 elems/thread.
__global__ __launch_bounds__(256) void ln_kernel(
    const float* __restrict__ in, const float* __restrict__ w,
    const float* __restrict__ b, float* __restrict__ out, int D)
{
    __shared__ float red[256];
    int row = blockIdx.x, tid = threadIdx.x;
    const float* rp = in + (size_t)row * D;
    float v0 = rp[tid], v1 = rp[tid + 256];
    red[tid] = v0 + v1; __syncthreads();
    for (int s = 128; s > 0; s >>= 1) { if (tid < s) red[tid] += red[tid + s]; __syncthreads(); }
    float mean = red[0] / D;
    __syncthreads();
    float d0 = v0 - mean, d1 = v1 - mean;
    red[tid] = d0 * d0 + d1 * d1; __syncthreads();
    for (int s = 128; s > 0; s >>= 1) { if (tid < s) red[tid] += red[tid + s]; __syncthreads(); }
    float rstd = rsqrtf(red[0] / D + 1e-5f);
    float* op = out + (size_t)row * D;
    op[tid]       = d0 * rstd * w[tid]       + b[tid];
    op[tid + 256] = d1 * rstd * w[tid + 256] + b[tid + 256];
}

// token-shift mix -> bf16 operands (xk/xv/xr); null outputs skipped.
__global__ void mix_kernel(const float* __restrict__ xn,
    const float* __restrict__ mk, const float* __restrict__ mv,
    const float* __restrict__ mr,
    bf16* __restrict__ xk, bf16* __restrict__ xv, bf16* __restrict__ xr,
    int T, int D)
{
    size_t i = (size_t)blockIdx.x * blockDim.x + threadIdx.x;
    int d = (int)(i % D);
    size_t row = i / D;
    int t = (int)(row % T);
    float cur  = xn[i];
    float prev = (t == 0) ? 0.0f : xn[i - D];
    if (xk) { float m = mk[d]; xk[i] = (bf16)(cur * m + prev * (1.0f - m)); }
    if (xv) { float m = mv[d]; xv[i] = (bf16)(cur * m + prev * (1.0f - m)); }
    if (xr) { float m = mr[d]; xr[i] = (bf16)(cur * m + prev * (1.0f - m)); }
}

// f32 -> bf16 elementwise
__global__ void f2bf_kernel(const float* __restrict__ in, bf16* __restrict__ out)
{
    size_t i = (size_t)blockIdx.x * blockDim.x + threadIdx.x;
    out[i] = (bf16)in[i];
}

// fp32 [L][K][N] -> bf16 [L][N][K] (transpose per matrix) for WMMA B-operand.
__global__ void wtrans_kernel(const float* __restrict__ W, bf16* __restrict__ Wt,
                              int K, int N, size_t total)
{
    size_t i = (size_t)blockIdx.x * blockDim.x + threadIdx.x;
    if (i >= total) return;
    size_t KN = (size_t)K * N;
    size_t l = i / KN, j = i % KN;
    int k = (int)(j % K);
    int n = (int)(j / K);
    Wt[i] = (bf16)W[l * KN + (size_t)k * N + n];
}

// ---------------------------------------------------------------------------
static inline void launch_gemm(const bf16* A, const bf16* Wt, float* C, bf16* Cbf,
                               const float* gate, int M, int N, int K, int flags,
                               hipStream_t s)
{
    dim3 grid(N / BN, M / BM);
    gemm_bf16_kernel<<<grid, 256, 0, s>>>(A, Wt, C, Cbf, gate, M, N, K, flags);
}

extern "C" void kernel_launch(void* const* d_in, const int* in_sizes, int n_in,
                              void* d_out, int out_size, void* d_ws, size_t ws_size,
                              hipStream_t stream)
{
    const int Bv = 4, Tv = 2048, Ev = 4, Dv = 512, Fv = 2048, Lv = 8, OUTv = 512;
    const int M = Bv * Tv;   // 8192 token rows

    const float* x       = (const float*)d_in[0];
    const float* emb_w   = (const float*)d_in[1];
    const float* emb_b   = (const float*)d_in[2];
    const float* ln_in_w = (const float*)d_in[3];
    const float* ln_in_b = (const float*)d_in[4];
    const float* ln0_w   = (const float*)d_in[5];
    const float* ln0_b   = (const float*)d_in[6];
    const float* ln1_w   = (const float*)d_in[7];
    const float* ln1_b   = (const float*)d_in[8];
    const float* tm_mix_k = (const float*)d_in[9];
    const float* tm_mix_v = (const float*)d_in[10];
    const float* tm_mix_r = (const float*)d_in[11];
    const float* tm_decay = (const float*)d_in[12];
    const float* tm_first = (const float*)d_in[13];
    const float* tm_wk   = (const float*)d_in[14];
    const float* tm_wv   = (const float*)d_in[15];
    const float* tm_wr   = (const float*)d_in[16];
    const float* tm_wo   = (const float*)d_in[17];
    const float* cm_mix_k = (const float*)d_in[18];
    const float* cm_mix_r = (const float*)d_in[19];
    const float* cm_wk   = (const float*)d_in[20];
    const float* cm_wv   = (const float*)d_in[21];
    const float* cm_wr   = (const float*)d_in[22];
    const float* ln_out_w = (const float*)d_in[23];
    const float* ln_out_b = (const float*)d_in[24];
    const float* head_w  = (const float*)d_in[25];
    float* out = (float*)d_out;

    // ---- bump-allocate workspace ----
    char* wp = (char*)d_ws;
    auto alloc = [&](size_t bytes) -> void* {
        void* p = (void*)wp;
        wp += (bytes + 255) & ~(size_t)255;
        return p;
    };
    float* h    = (float*)alloc((size_t)M * Dv * 4);
    float* xn   = (float*)alloc((size_t)M * Dv * 4);
    float* kbuf = (float*)alloc((size_t)M * Dv * 4);
    float* vbuf = (float*)alloc((size_t)M * Dv * 4);
    float* rbuf = (float*)alloc((size_t)M * Dv * 4);   // also reused as rf gate
    bf16* xk_bf   = (bf16*)alloc((size_t)M * Dv * 2);
    bf16* xv_bf   = (bf16*)alloc((size_t)M * Dv * 2);
    bf16* xr_bf   = (bf16*)alloc((size_t)M * Dv * 2);
    bf16* rwkv_bf = (bf16*)alloc((size_t)M * Dv * 2);
    bf16* kf_bf   = (bf16*)alloc((size_t)M * Fv * 2);
    bf16* tm_wkT = (bf16*)alloc((size_t)Lv * Dv * Dv * 2);
    bf16* tm_wvT = (bf16*)alloc((size_t)Lv * Dv * Dv * 2);
    bf16* tm_wrT = (bf16*)alloc((size_t)Lv * Dv * Dv * 2);
    bf16* tm_woT = (bf16*)alloc((size_t)Lv * Dv * Dv * 2);
    bf16* cm_wkT = (bf16*)alloc((size_t)Lv * Dv * Fv * 2);
    bf16* cm_wvT = (bf16*)alloc((size_t)Lv * Fv * Dv * 2);
    bf16* cm_wrT = (bf16*)alloc((size_t)Lv * Dv * Dv * 2);
    bf16* headT  = (bf16*)alloc((size_t)Dv * OUTv * 2);

    // ---- weight conversion + transpose to bf16 [N][K] ----
    {
        size_t tDD = (size_t)Lv * Dv * Dv;
        int g = (int)((tDD + 255) / 256);
        wtrans_kernel<<<g, 256, 0, stream>>>(tm_wk, tm_wkT, Dv, Dv, tDD);
        wtrans_kernel<<<g, 256, 0, stream>>>(tm_wv, tm_wvT, Dv, Dv, tDD);
        wtrans_kernel<<<g, 256, 0, stream>>>(tm_wr, tm_wrT, Dv, Dv, tDD);
        wtrans_kernel<<<g, 256, 0, stream>>>(tm_wo, tm_woT, Dv, Dv, tDD);
        wtrans_kernel<<<g, 256, 0, stream>>>(cm_wr, cm_wrT, Dv, Dv, tDD);
        size_t tDF = (size_t)Lv * Dv * Fv;
        int g2 = (int)((tDF + 255) / 256);
        wtrans_kernel<<<g2, 256, 0, stream>>>(cm_wk, cm_wkT, Dv, Fv, tDF);
        wtrans_kernel<<<g2, 256, 0, stream>>>(cm_wv, cm_wvT, Fv, Dv, tDF);
        size_t tH = (size_t)Dv * OUTv;
        wtrans_kernel<<<(int)((tH + 255) / 256), 256, 0, stream>>>(head_w, headT, Dv, OUTv, tH);
    }

    // ---- embedding + input LN ----
    embed_ln_kernel<<<M, 256, 0, stream>>>(x, emb_w, emb_b, ln_in_w, ln_in_b, h, Ev, Dv);

    const int mixGrid = (int)(((size_t)M * Dv) / 256);

    for (int i = 0; i < Lv; ++i) {
        size_t oD  = (size_t)i * Dv;
        size_t oDD = (size_t)i * Dv * Dv;
        size_t oDF = (size_t)i * Dv * Fv;

        // ---- time mixing ----
        ln_kernel<<<M, 256, 0, stream>>>(h, ln0_w + oD, ln0_b + oD, xn, Dv);
        mix_kernel<<<mixGrid, 256, 0, stream>>>(xn, tm_mix_k + oD, tm_mix_v + oD,
                                                tm_mix_r + oD, xk_bf, xv_bf, xr_bf, Tv, Dv);
        launch_gemm(xk_bf, tm_wkT + oDD, kbuf, nullptr, nullptr, M, Dv, Dv, EPI_F32, stream);
        launch_gemm(xv_bf, tm_wvT + oDD, vbuf, nullptr, nullptr, M, Dv, Dv, EPI_F32, stream);
        launch_gemm(xr_bf, tm_wrT + oDD, rbuf, nullptr, nullptr, M, Dv, Dv, EPI_F32 | EPI_SIG, stream);
        wkv_kernel<<<(Bv * Dv) / 256, 256, 0, stream>>>(kbuf, vbuf, rbuf,
                                                        tm_decay + oD, tm_first + oD,
                                                        rwkv_bf, Tv, Dv);
        launch_gemm(rwkv_bf, tm_woT + oDD, h, nullptr, nullptr, M, Dv, Dv, EPI_ADD, stream);

        // ---- channel mixing ----
        ln_kernel<<<M, 256, 0, stream>>>(h, ln1_w + oD, ln1_b + oD, xn, Dv);
        mix_kernel<<<mixGrid, 256, 0, stream>>>(xn, cm_mix_k + oD, nullptr,
                                                cm_mix_r + oD, xk_bf, nullptr, xr_bf, Tv, Dv);
        launch_gemm(xk_bf, cm_wkT + oDF, h /*unused*/, kf_bf, nullptr,
                    M, Fv, Dv, EPI_RELUSQ | EPI_BF16, stream);
        launch_gemm(xr_bf, cm_wrT + oDD, rbuf, nullptr, nullptr, M, Dv, Dv,
                    EPI_F32 | EPI_SIG, stream);
        launch_gemm(kf_bf, cm_wvT + oDF, h, nullptr, rbuf, M, Dv, Fv,
                    EPI_ADD | EPI_GATE, stream);
    }

    // ---- output LN + head ----
    ln_kernel<<<M, 256, 0, stream>>>(h, ln_out_w, ln_out_b, xn, Dv);
    f2bf_kernel<<<mixGrid, 256, 0, stream>>>(xn, xk_bf);
    launch_gemm(xk_bf, headT, out, nullptr, nullptr, M, OUTv, Dv, EPI_F32, stream);
}
